// ConvLayer_50869592655523
// MI455X (gfx1250) — compile-verified, matrix operand order checked by reading, not run on hardware
//
#include <hip/hip_runtime.h>
#include <hip/hip_bf16.h>

// ---------------------------------------------------------------------------
// CDNA5 / gfx1250 implementation.
// Dense MLP layers: v_wmma_f32_16x16x32_f16 (f16 A/B, f32 acc). Block-shared
// B fragments staged into LDS via global_load_async_to_lds_b128 (ASYNCcnt),
// consumed via ds_load_b128 with software-pipelined (double-buffered) A/B
// fragments so the 4-WMMA chain per K-tile issues without per-strip waits.
// Segment-sums: global_atomic_add_f32.
// ---------------------------------------------------------------------------

typedef __attribute__((ext_vector_type(16))) _Float16 v16h;
typedef __attribute__((ext_vector_type(8)))  float    v8f;

#define D_       128
#define N_NODES_ 100000
#define N_EDGES_ 200000
#define C_ROWS_  240000
#define N_CYC_   40000
#define B_ROWS_  140000
#define N_BR_    20000
#define EN_      400000
#define NE_      400000
#define EC_      480000
#define CE_      480000
#define EB_      280000
#define BE_      280000

struct Piece {
  const float*    A;    // [M_src, lda] fp32 row-major, 128 K-columns used
  const int*      idx;  // optional row gather (c_sum / b_sum broadcast)
  const _Float16* W;    // packed B fragments for this 128-row weight slice
  int             lda;
};

struct GemmArgs {
  Piece        p[4];
  const float* bias;   // [N]
  float*       out;    // [M, N] row-major
  int M, N, npieces, relu;
};

// Build the f16 A-fragment for a 16x32 K-tile per the CDNA5 16-bit A layout:
// lane p (p16 = p&15 is the row), halves 0..7 = K kb..kb+7, 8..15 = K kb+16..kb+23,
// where kb already includes the hi-half (+8) shift for lanes 16..31.
__device__ __forceinline__ v16h load_a_frag(const float* ar, int kb) {
  float4 x0 = *(const float4*)(ar + kb);
  float4 x1 = *(const float4*)(ar + kb + 4);
  float4 x2 = *(const float4*)(ar + kb + 16);
  float4 x3 = *(const float4*)(ar + kb + 20);
  v16h a;
  a[0]  = (_Float16)x0.x; a[1]  = (_Float16)x0.y; a[2]  = (_Float16)x0.z; a[3]  = (_Float16)x0.w;
  a[4]  = (_Float16)x1.x; a[5]  = (_Float16)x1.y; a[6]  = (_Float16)x1.z; a[7]  = (_Float16)x1.w;
  a[8]  = (_Float16)x2.x; a[9]  = (_Float16)x2.y; a[10] = (_Float16)x2.z; a[11] = (_Float16)x2.w;
  a[12] = (_Float16)x3.x; a[13] = (_Float16)x3.y; a[14] = (_Float16)x3.z; a[15] = (_Float16)x3.w;
  return a;
}

// LDS B-fragment fetch: piece-local chunk (kt,strip), one 32B v16h per lane.
__device__ __forceinline__ v16h ldsb(const _Float16* sbase, int kt, int s, int lane) {
  return *((const v16h*)(sbase + kt * 2048) + s * 32 + lane);
}

// One wave computes a 16(M) x 64(N) output strip; 8 waves per block -> 128 rows.
// K is consumed in 32-wide tiles; each concat "piece" contributes K=128 (4 tiles).
// All 8 waves share the same 64 output columns, so the B fragments for the whole
// block (npieces * 16KB) are staged into LDS once via async copy.
__global__ __launch_bounds__(256) void mlp_wmma_gemm(GemmArgs g) {
  __shared__ _Float16 sB[4 * 16 * 512];        // [piece][kt*4+strip][lane*16+j], 64KB
  const int tid   = threadIdx.x;
  const int lane  = tid & 31;
  const int wave  = tid >> 5;
  const int mtile = blockIdx.x * 8 + wave;
  const int nbase  = blockIdx.y * 64;
  const int ntiles = g.N >> 4;
  const int nt0    = nbase >> 4;

  // ---- async-stage this block's B fragments (done by ALL waves, incl. tail)
  for (int p = 0; p < g.npieces; ++p) {
    const _Float16* src = g.p[p].W;
    for (int u = tid; u < 1024; u += 256) {          // 1024 x 16B units per piece
      const int c  = u >> 6;                         // kt*4 + strip (0..15)
      const int w  = u & 63;                         // 16B unit within 1KB chunk
      const int kt = c >> 2, st = c & 3;
      const _Float16* gp = src + (size_t)(kt * ntiles + nt0 + st) * 512 + w * 8;
      unsigned lds_off = (unsigned)(unsigned long long)&sB[(p * 16 + c) * 512 + w * 8];
      unsigned long long ga = (unsigned long long)gp;
      asm volatile("global_load_async_to_lds_b128 %0, %1, off"
                   :: "v"(lds_off), "v"(ga) : "memory");
    }
  }
  asm volatile("s_wait_asynccnt 0x0" ::: "memory");
  __syncthreads();

  if (mtile * 16 >= g.M) return;               // wave-uniform exit (EXEC all-1 for WMMA)

  const int p16 = lane & 15;
  const int hi  = lane >> 4;
  v8f acc[4] = {};

  for (int pi = 0; pi < g.npieces; ++pi) {
    const Piece pc = g.p[pi];
    int row = mtile * 16 + p16;
    if (pc.idx) row = pc.idx[row];
    const float* ar = pc.A + (size_t)row * pc.lda;
    const _Float16* sbase = &sB[pi * 16 * 512];

    // software pipeline: prefetch K-tile kt+1 (A via global+cvt, B via 8x
    // ds_load_b128 into a second buffer) before the 4-WMMA chain of tile kt.
    v16h a_c  = load_a_frag(ar, hi * 8);
    v16h b_c0 = ldsb(sbase, 0, 0, lane);
    v16h b_c1 = ldsb(sbase, 0, 1, lane);
    v16h b_c2 = ldsb(sbase, 0, 2, lane);
    v16h b_c3 = ldsb(sbase, 0, 3, lane);
#pragma unroll
    for (int kt = 0; kt < 4; ++kt) {
      const v16h a  = a_c;
      const v16h b0 = b_c0, b1 = b_c1, b2 = b_c2, b3 = b_c3;
      if (kt < 3) {
        a_c  = load_a_frag(ar, (kt + 1) * 32 + hi * 8);
        b_c0 = ldsb(sbase, kt + 1, 0, lane);
        b_c1 = ldsb(sbase, kt + 1, 1, lane);
        b_c2 = ldsb(sbase, kt + 1, 2, lane);
        b_c3 = ldsb(sbase, kt + 1, 3, lane);
      }
      acc[0] = __builtin_amdgcn_wmma_f32_16x16x32_f16(false, a, false, b0,
                                                      (short)0, acc[0], false, false);
      acc[1] = __builtin_amdgcn_wmma_f32_16x16x32_f16(false, a, false, b1,
                                                      (short)0, acc[1], false, false);
      acc[2] = __builtin_amdgcn_wmma_f32_16x16x32_f16(false, a, false, b2,
                                                      (short)0, acc[2], false, false);
      acc[3] = __builtin_amdgcn_wmma_f32_16x16x32_f16(false, a, false, b3,
                                                      (short)0, acc[3], false, false);
    }
  }

  // C/D layout: VGPR r -> row mtile*16 + hi*8 + r, col nbase + strip*16 + (lane&15)
  const size_t rb = (size_t)(mtile * 16 + hi * 8);
#pragma unroll
  for (int s = 0; s < 4; ++s) {
    const int col = nbase + s * 16 + p16;
    const float bv = g.bias[col];
#pragma unroll
    for (int r = 0; r < 8; ++r) {
      float v = acc[s][r] + bv;
      if (g.relu) v = fmaxf(v, 0.0f);
      g.out[(rb + r) * (size_t)g.N + col] = v;
    }
  }
}

// Pack a 128-row slice of fp32 weight W[rowOff..rowOff+128, 0..N) (row stride ldw)
// into contiguous per-lane B fragments: out[((kt*ntiles+nt)*32 + lane)*16 + j].
__global__ void pack_w_f16(const float* __restrict__ W, int ldw, int rowOff, int N,
                           _Float16* __restrict__ out) {
  int t = blockIdx.x * blockDim.x + threadIdx.x;
  if (t >= 128 * N) return;
  int j    = t & 15;
  int lane = (t >> 4) & 31;
  int rest = t >> 9;
  int ntiles = N >> 4;
  int nt = rest % ntiles;
  int kt = rest / ntiles;
  int k = kt * 32 + (j < 8 ? j : 8 + j) + ((lane >= 16) ? 8 : 0);
  int n = nt * 16 + (lane & 15);
  out[t] = (_Float16)W[(size_t)(rowOff + k) * ldw + n];
}

__global__ void zero_f32(float* p, size_t n) {
  size_t i = ((size_t)blockIdx.x * blockDim.x + threadIdx.x) * 4;
  if (i < n) *(float4*)(p + i) = make_float4(0.f, 0.f, 0.f, 0.f);
}

// out[dst[e]] += in[src?src[e]:e], 128 cols; one wave per index entry.
__global__ void scatter_add128(const float* __restrict__ in, const int* __restrict__ src,
                               const int* __restrict__ dst, int n, float* __restrict__ out) {
  long long t = (long long)blockIdx.x * blockDim.x + threadIdx.x;
  int e = (int)(t >> 5);
  if (e >= n) return;
  int lane = (int)(t & 31);
  int s = src ? src[e] : e;
  float4 v = *(const float4*)(in + (size_t)s * D_ + lane * 4);
  float* o = out + (size_t)dst[e] * D_ + lane * 4;
  atomicAdd(o + 0, v.x); atomicAdd(o + 1, v.y);
  atomicAdd(o + 2, v.z); atomicAdd(o + 3, v.w);
}

// W_comb[128,256] = W_aut[128,512] @ W_br0[384:896, 0:256]  (Autobahn fold)
__global__ void wcomb_gemm(const float* __restrict__ Waut, const float* __restrict__ Wbr0,
                           float* __restrict__ out) {
  int t = blockIdx.x * blockDim.x + threadIdx.x;
  if (t >= 128 * 256) return;
  int i = t >> 8, j = t & 255;
  float s = 0.f;
  for (int k = 0; k < 512; ++k) s += Waut[i * 512 + k] * Wbr0[(size_t)(384 + k) * 256 + j];
  out[t] = s;
}

// bias_extra[j] = b_br0[j] + sum_k b_aut[k] * W_br0[384+k, j]
__global__ void bextra_k(const float* __restrict__ baut, const float* __restrict__ Wbr0,
                         const float* __restrict__ b0, float* __restrict__ out) {
  int j = blockIdx.x * blockDim.x + threadIdx.x;
  if (j >= 256) return;
  float s = b0[j];
  for (int k = 0; k < 512; ++k) s += baut[k] * Wbr0[(size_t)(384 + k) * 256 + j];
  out[j] = s;
}

// ------------------------------- host side ---------------------------------

static const size_t WSLOT = 128 * 256; // halves per packed weight slot

static void packW(hipStream_t s, const float* W, int ldw, int rowOff, int N, _Float16* dst) {
  int tot = 128 * N;
  pack_w_f16<<<(tot + 255) / 256, 256, 0, s>>>(W, ldw, rowOff, N, dst);
}

static void gemm(hipStream_t s, int M, int N, int np, const Piece* ps,
                 const float* bias, float* out, int relu) {
  GemmArgs g{};
  for (int i = 0; i < np; ++i) g.p[i] = ps[i];
  g.bias = bias; g.out = out; g.M = M; g.N = N; g.npieces = np; g.relu = relu;
  dim3 grid((unsigned)((M + 127) / 128), (unsigned)(N / 64));
  mlp_wmma_gemm<<<grid, 256, 0, s>>>(g);
}

static void zero(hipStream_t s, float* p, size_t n) {
  zero_f32<<<(unsigned)((n / 4 + 255) / 256), 256, 0, s>>>(p, n);
}

static void scat(hipStream_t s, const float* in, const int* src, const int* dst,
                 int n, float* out) {
  long long tot = (long long)n * 32;
  scatter_add128<<<(unsigned)((tot + 255) / 256), 256, 0, s>>>(in, src, dst, n, out);
}

// 2-piece-input, 2-layer MLP: out = ( relu([x0|x1]@W0+b0) ) @ W1 + b1
static void mlp2(hipStream_t s, int M, const float* x0, const float* x1,
                 const float* W0, const float* b0, const float* W1, const float* b1,
                 float* H1, _Float16* wpk, float* out) {
  packW(s, W0, 256, 0,   256, wpk + 0 * WSLOT);
  packW(s, W0, 256, 128, 256, wpk + 1 * WSLOT);
  Piece p1[2] = {{x0, nullptr, wpk + 0 * WSLOT, 128}, {x1, nullptr, wpk + 1 * WSLOT, 128}};
  gemm(s, M, 256, 2, p1, b0, H1, 1);
  packW(s, W1, 128, 0,   128, wpk + 2 * WSLOT);
  packW(s, W1, 128, 128, 128, wpk + 3 * WSLOT);
  Piece p2[2] = {{H1, nullptr, wpk + 2 * WSLOT, 256}, {H1 + 128, nullptr, wpk + 3 * WSLOT, 256}};
  gemm(s, M, 128, 2, p2, b1, out, 0);
}

extern "C" void kernel_launch(void* const* d_in, const int* in_sizes, int n_in,
                              void* d_out, int out_size, void* d_ws, size_t ws_size,
                              hipStream_t stream) {
  (void)in_sizes; (void)n_in; (void)out_size; (void)ws_size;

  const float* node_rep  = (const float*)d_in[0];
  const float* edge_rep  = (const float*)d_in[1];
  const float* cycle_rep = (const float*)d_in[2];
  const float* br_rep    = (const float*)d_in[3];
  const int* en_src = (const int*)d_in[4];  const int* en_dst = (const int*)d_in[5];
  const int* ne_src = (const int*)d_in[6];  const int* ne_dst = (const int*)d_in[7];
  const int* ec_src = (const int*)d_in[8];  const int* ec_dst = (const int*)d_in[9];
  const int* cyc_seg = (const int*)d_in[10];
  const int* ce_src = (const int*)d_in[11]; const int* ce_dst = (const int*)d_in[12];
  const int* eb_src = (const int*)d_in[13]; const int* eb_dst = (const int*)d_in[14];
  const int* br_seg = (const int*)d_in[15];
  const int* be_src = (const int*)d_in[16]; const int* be_dst = (const int*)d_in[17];
  const float* We0_0 = (const float*)d_in[18]; const float* be0_0 = (const float*)d_in[19];
  const float* We0_1 = (const float*)d_in[20]; const float* be0_1 = (const float*)d_in[21];
  const float* Wn_0  = (const float*)d_in[22]; const float* bn_0  = (const float*)d_in[23];
  const float* Wn_1  = (const float*)d_in[24]; const float* bn_1  = (const float*)d_in[25];
  const float* Wc_0  = (const float*)d_in[26]; const float* bc_0  = (const float*)d_in[27];
  const float* Wc_1  = (const float*)d_in[28]; const float* bc_1  = (const float*)d_in[29];
  const float* Wc_2  = (const float*)d_in[30]; const float* bc_2  = (const float*)d_in[31];
  const float* Wec_0 = (const float*)d_in[32]; const float* bec_0 = (const float*)d_in[33];
  const float* Wec_1 = (const float*)d_in[34]; const float* bec_1 = (const float*)d_in[35];
  const float* W_aut = (const float*)d_in[36]; const float* b_aut = (const float*)d_in[37];
  const float* Wb_0  = (const float*)d_in[38]; const float* bb_0  = (const float*)d_in[39];
  const float* Wb_1  = (const float*)d_in[40]; const float* bb_1  = (const float*)d_in[41];
  const float* Wb_2  = (const float*)d_in[42]; const float* bb_2  = (const float*)d_in[43];
  const float* Wbc_0 = (const float*)d_in[44]; const float* bbc_0 = (const float*)d_in[45];
  const float* Wbc_1 = (const float*)d_in[46]; const float* bbc_1 = (const float*)d_in[47];
  const float* Wf_0  = (const float*)d_in[48]; const float* bf_0  = (const float*)d_in[49];
  const float* Wf_1  = (const float*)d_in[50]; const float* bf_1  = (const float*)d_in[51];
  const float* Wf_2  = (const float*)d_in[52]; const float* bf_2  = (const float*)d_in[53];

  float* out_node  = (float*)d_out;
  float* out_edge  = out_node  + (size_t)N_NODES_ * D_;
  float* out_cycle = out_edge  + (size_t)N_EDGES_ * D_;
  float* out_br    = out_cycle + (size_t)C_ROWS_ * D_;

  // ---- workspace carving (fp32 units, 256B aligned slots)
  size_t off = 0;
  float* wsf = (float*)d_ws;
  auto carve = [&](size_t nflts) { float* p = wsf + off; off += (nflts + 63) & ~(size_t)63; return p; };
  float* H1 = carve((size_t)C_ROWS_ * 256);   // largest hidden
  float* H2 = carve((size_t)C_ROWS_ * 256);
  float* A  = carve((size_t)N_EDGES_ * D_);   // node2edge / edge2node / c2e / b2e
  float* C  = carve((size_t)C_ROWS_ * D_);    // e2c / e2b
  float* Dp = carve((size_t)N_CYC_ * D_);     // cycsum / brsum
  float* E1 = carve((size_t)N_EDGES_ * D_);
  float* E2 = carve((size_t)N_EDGES_ * D_);
  float* E3 = carve((size_t)N_EDGES_ * D_);
  float* WCOMB = carve(128 * 256);
  float* BEX   = carve(256);
  _Float16* wpk = (_Float16*)carve(8 * WSLOT / 2 + 64);

  // ======================= Edge_node =======================
  zero(stream, A, (size_t)N_EDGES_ * D_);
  scat(stream, node_rep, en_src, en_dst, EN_, A);
  mlp2(stream, N_EDGES_, edge_rep, A, We0_0, be0_0, We0_1, be0_1, H1, wpk, E1);

  zero(stream, A, (size_t)N_NODES_ * D_);
  scat(stream, E1, ne_src, ne_dst, NE_, A);
  mlp2(stream, N_NODES_, node_rep, A, Wn_0, bn_0, Wn_1, bn_1, H1, wpk, out_node);

  // ======================= Edge_cycle ======================
  zero(stream, C, (size_t)C_ROWS_ * D_);
  scat(stream, edge_rep, ec_src, ec_dst, EC_, C);
  zero(stream, Dp, (size_t)N_CYC_ * D_);
  scat(stream, C, nullptr, cyc_seg, C_ROWS_, Dp);
  {
    packW(stream, Wc_0, 256, 0,   256, wpk + 0 * WSLOT);
    packW(stream, Wc_0, 256, 128, 256, wpk + 1 * WSLOT);
    packW(stream, Wc_0, 256, 256, 256, wpk + 2 * WSLOT);
    Piece p1[3] = {{cycle_rep, nullptr, wpk + 0 * WSLOT, 128},
                   {C,         nullptr, wpk + 1 * WSLOT, 128},
                   {Dp,        cyc_seg, wpk + 2 * WSLOT, 128}};
    gemm(stream, C_ROWS_, 256, 3, p1, bc_0, H1, 1);
    packW(stream, Wc_1, 256, 0,   256, wpk + 3 * WSLOT);
    packW(stream, Wc_1, 256, 128, 256, wpk + 4 * WSLOT);
    Piece p2[2] = {{H1, nullptr, wpk + 3 * WSLOT, 256}, {H1 + 128, nullptr, wpk + 4 * WSLOT, 256}};
    gemm(stream, C_ROWS_, 256, 2, p2, bc_1, H2, 1);
    packW(stream, Wc_2, 128, 0,   128, wpk + 5 * WSLOT);
    packW(stream, Wc_2, 128, 128, 128, wpk + 6 * WSLOT);
    Piece p3[2] = {{H2, nullptr, wpk + 5 * WSLOT, 256}, {H2 + 128, nullptr, wpk + 6 * WSLOT, 256}};
    gemm(stream, C_ROWS_, 128, 2, p3, bc_2, out_cycle, 0);
  }
  zero(stream, A, (size_t)N_EDGES_ * D_);
  scat(stream, out_cycle, ce_src, ce_dst, CE_, A);
  mlp2(stream, N_EDGES_, edge_rep, A, Wec_0, bec_0, Wec_1, bec_1, H1, wpk, E2);

  // ================== Edge_branched_cycle ==================
  zero(stream, C, (size_t)B_ROWS_ * D_);
  scat(stream, edge_rep, eb_src, eb_dst, EB_, C);
  zero(stream, Dp, (size_t)N_BR_ * D_);
  scat(stream, C, nullptr, br_seg, B_ROWS_, Dp);
  wcomb_gemm<<<(128 * 256 + 255) / 256, 256, 0, stream>>>(W_aut, Wb_0, WCOMB);
  bextra_k<<<1, 256, 0, stream>>>(b_aut, Wb_0, bb_0, BEX);
  {
    packW(stream, Wb_0, 256, 0,   256, wpk + 0 * WSLOT);
    packW(stream, Wb_0, 256, 128, 256, wpk + 1 * WSLOT);
    packW(stream, Wb_0, 256, 256, 256, wpk + 2 * WSLOT);
    packW(stream, WCOMB, 256, 0,  256, wpk + 3 * WSLOT);   // folded Autobahn term
    Piece p1[4] = {{br_rep, nullptr, wpk + 0 * WSLOT, 128},
                   {C,      nullptr, wpk + 1 * WSLOT, 128},
                   {Dp,     br_seg,  wpk + 2 * WSLOT, 128},
                   {C,      nullptr, wpk + 3 * WSLOT, 128}};
    gemm(stream, B_ROWS_, 256, 4, p1, BEX, H1, 1);
    packW(stream, Wb_1, 256, 0,   256, wpk + 4 * WSLOT);
    packW(stream, Wb_1, 256, 128, 256, wpk + 5 * WSLOT);
    Piece p2[2] = {{H1, nullptr, wpk + 4 * WSLOT, 256}, {H1 + 128, nullptr, wpk + 5 * WSLOT, 256}};
    gemm(stream, B_ROWS_, 256, 2, p2, bb_1, H2, 1);
    packW(stream, Wb_2, 128, 0,   128, wpk + 6 * WSLOT);
    packW(stream, Wb_2, 128, 128, 128, wpk + 7 * WSLOT);
    Piece p3[2] = {{H2, nullptr, wpk + 6 * WSLOT, 256}, {H2 + 128, nullptr, wpk + 7 * WSLOT, 256}};
    gemm(stream, B_ROWS_, 128, 2, p3, bb_2, out_br, 0);
  }
  zero(stream, A, (size_t)N_EDGES_ * D_);
  scat(stream, out_br, be_src, be_dst, BE_, A);
  mlp2(stream, N_EDGES_, edge_rep, A, Wbc_0, bbc_0, Wbc_1, bbc_1, H1, wpk, E3);

  // ===================== final edge MLP ====================
  {
    packW(stream, Wf_0, 256, 0,   256, wpk + 0 * WSLOT);
    packW(stream, Wf_0, 256, 128, 256, wpk + 1 * WSLOT);
    packW(stream, Wf_0, 256, 256, 256, wpk + 2 * WSLOT);
    Piece p1[3] = {{E1, nullptr, wpk + 0 * WSLOT, 128},
                   {E2, nullptr, wpk + 1 * WSLOT, 128},
                   {E3, nullptr, wpk + 2 * WSLOT, 128}};
    gemm(stream, N_EDGES_, 256, 3, p1, bf_0, H1, 1);
    packW(stream, Wf_1, 256, 0,   256, wpk + 3 * WSLOT);
    packW(stream, Wf_1, 256, 128, 256, wpk + 4 * WSLOT);
    Piece p2[2] = {{H1, nullptr, wpk + 3 * WSLOT, 256}, {H1 + 128, nullptr, wpk + 4 * WSLOT, 256}};
    gemm(stream, N_EDGES_, 256, 2, p2, bf_1, H2, 1);
    packW(stream, Wf_2, 128, 0,   128, wpk + 5 * WSLOT);
    packW(stream, Wf_2, 128, 128, 128, wpk + 6 * WSLOT);
    Piece p3[2] = {{H2, nullptr, wpk + 5 * WSLOT, 256}, {H2 + 128, nullptr, wpk + 6 * WSLOT, 256}};
    gemm(stream, N_EDGES_, 128, 2, p3, bf_2, out_edge, 0);
  }
}
